// MambaASTGCN_block_2731599200630
// MI455X (gfx1250) — matrix operand order repaired and checked by reading, not compiled, not run on hardware
//
#include <hip/hip_runtime.h>
#include <math.h>

// Problem dims (from reference)
#define BB   16
#define NN   512
#define FIN  64
#define TT   24
#define OO   64
#define KCH  3      // Chebyshev order
#define DI   128
#define DS   16
#define DCONVW 4
#define DTR  4
#define EPSL 1e-5f

typedef __attribute__((ext_vector_type(16))) _Float16 v16h;
typedef __attribute__((ext_vector_type(8)))  _Float16 v8h;
typedef __attribute__((ext_vector_type(8)))  float    v8f;

// ---------------------------------------------------------------------------
// Generic 16x16 output-tile GEMM accumulator using V_WMMA_F32_16X16X32_F16
// with direct (functor) operand loads. Used only for the small-K GEMMs.
// Fragment layouts per CDNA5 ISA 7.12.2.
// ---------------------------------------------------------------------------
template<typename LA, typename LB>
__device__ inline v8f wmma_tile_f16(LA loadA, LB loadB, int Kdim, v8f acc) {
  const int lane = threadIdx.x & 31;
  const int half = lane >> 4;
  const int idx  = lane & 15;
  for (int k0 = 0; k0 < Kdim; k0 += 32) {
    v16h af, bf;
#pragma unroll
    for (int j = 0; j < 16; ++j) {
      const int v     = j >> 1;
      const int kbase = (v < 4) ? (2 * v) : (16 + 2 * (v - 4));
      af[j] = (_Float16)loadA(idx, k0 + kbase + 8 * half + (j & 1));
      bf[j] = (_Float16)loadB(k0 + kbase + 8 * half + (j & 1), idx);
    }
    acc = __builtin_amdgcn_wmma_f32_16x16x32_f16(false, af, false, bf,
                                                 (short)0, acc, false, false);
  }
  return acc;
}

// Build A fragment from LDS tile sA[64][32] (f16): one contiguous 16-half run (k = 16*half + s).
__device__ inline v16h frag_a_lds(const _Float16* sA /*row base*/, int half) {
  const v8h lo = *reinterpret_cast<const v8h*>(sA + 16 * half);
  const v8h hi = *reinterpret_cast<const v8h*>(sA + 16 * half + 8);
  return __builtin_shufflevector(lo, hi, 0, 1, 2, 3, 4, 5, 6, 7,
                                 8, 9, 10, 11, 12, 13, 14, 15);
}

// ---------------------------------------------------------------------------
// 1) lhs[b,n,t] = (sum_t' x*W1) @ W2 ; rhs[b,n,t] = sum_f W3[f]*x[b,n,f,t]
// ---------------------------------------------------------------------------
__global__ void __launch_bounds__(64)
k_lhs_rhs(const float* __restrict__ x, const float* __restrict__ W1,
          const float* __restrict__ W2, const float* __restrict__ W3,
          float* __restrict__ lhs, float* __restrict__ rhs) {
  const int n = blockIdx.x, b = blockIdx.y;
  const int f = threadIdx.x;                     // 0..63
  __shared__ float s_xw1[FIN];
  __shared__ float s_xw3[FIN][TT];
  const float* xr = x + (((size_t)b * NN + n) * FIN + f) * TT;
  const float w3 = W3[f];
  float acc = 0.f;
#pragma unroll
  for (int t = 0; t < TT; ++t) {
    const float v = xr[t];
    acc += v * W1[t];
    s_xw3[f][t] = v * w3;
  }
  s_xw1[f] = acc;
  __syncthreads();
  if (f < TT) {
    const int t = f;
    float l = 0.f, rr = 0.f;
    for (int ff = 0; ff < FIN; ++ff) {
      l  += s_xw1[ff] * W2[ff * TT + t];
      rr += s_xw3[ff][t];
    }
    lhs[((size_t)b * NN + n) * TT + t] = l;
    rhs[((size_t)b * NN + n) * TT + t] = rr;
  }
}

// ---------------------------------------------------------------------------
// 2) sig[b,n,m] = sigmoid( sum_t lhs[b,n,t]*rhs[b,m,t] + bsp[n,m] )
//    K = 24 (one padded WMMA step) -> direct loads are fine.
// ---------------------------------------------------------------------------
__global__ void __launch_bounds__(32)
k_prod_sig(const float* __restrict__ lhs, const float* __restrict__ rhs,
           const float* __restrict__ bsp, float* __restrict__ sig) {
  const int m0 = blockIdx.x * 16, n0 = blockIdx.y * 16, b = blockIdx.z;
  const float* L = lhs + (size_t)b * NN * TT;
  const float* R = rhs + (size_t)b * NN * TT;
  v8f acc = {};
  acc = wmma_tile_f16(
      [&](int m, int k) { return (k < TT) ? L[(n0 + m) * TT + k] : 0.f; },
      [&](int k, int n) { return (k < TT) ? R[(m0 + n) * TT + k] : 0.f; },
      32, acc);
  const int lane = threadIdx.x & 31, half = lane >> 4, idx = lane & 15;
  float* Sg = sig + (size_t)b * NN * NN;
#pragma unroll
  for (int r = 0; r < 8; ++r) {
    const int row = n0 + r + 8 * half, col = m0 + idx;
    const float p = acc[r] + bsp[row * NN + col];
    Sg[(size_t)row * NN + col] = 1.f / (1.f + expf(-p));
  }
}

// ---------------------------------------------------------------------------
// 3) S[b,n,c] = sum_m Vs[n,m] * sig[b,m,c]       (K = 512)
//    8-wave block; 64x32 macro-tile; LDS-staged f16 operands; one WMMA/wave
//    per K-step read via ds_load_b128.
// ---------------------------------------------------------------------------
__global__ void __launch_bounds__(256)
k_Sgemm(const float* __restrict__ Vs, const float* __restrict__ sig,
        float* __restrict__ S) {
  const int c0 = blockIdx.x * 32, n0 = blockIdx.y * 64, b = blockIdx.z;
  const int tid  = threadIdx.x;
  const int wave = tid >> 5, lane = tid & 31;
  const int half = lane >> 4, idx = lane & 15;
  const int wr = wave >> 1, wc = wave & 1;       // 4x2 subtiles of 16x16

  __shared__ _Float16 sA[64][32];                // rows n, cols k (4 KB)
  __shared__ _Float16 sBt[32][32];               // rows n(col), cols k (2 KB)

  const float* Sg = sig + (size_t)b * NN * NN;
  v8f acc = {};

  for (int k0 = 0; k0 < NN; k0 += 32) {
    // Stage A: Vs rows n0..n0+63, k0..k0+31 (coalesced float2 reads)
    for (int it = tid; it < 64 * 16; it += 256) {
      const int m = it >> 4, kp = it & 15;
      const float* src = Vs + (size_t)(n0 + m) * NN + k0 + kp * 2;
      const float2 v = *reinterpret_cast<const float2*>(src);
      sA[m][kp * 2 + 0] = (_Float16)v.x;
      sA[m][kp * 2 + 1] = (_Float16)v.y;
      if (k0 + 32 < NN) __builtin_prefetch(src + 32, 0, 1);
    }
    // Stage B^T: sig rows k0..k0+31, cols c0..c0+31 (coalesced reads)
    for (int it = tid; it < 32 * 32; it += 256) {
      const int k = it >> 5, n = it & 31;
      const float* src = Sg + (size_t)(k0 + k) * NN + c0 + n;
      sBt[n][k] = (_Float16)(*src);
      if (k0 + 32 < NN) __builtin_prefetch(src + (size_t)32 * NN, 0, 1);
    }
    __syncthreads();
    const v16h af = frag_a_lds(&sA[wr * 16 + idx][0], half);
    const v16h bf = *reinterpret_cast<const v16h*>(&sBt[wc * 16 + idx][16 * half]);
    acc = __builtin_amdgcn_wmma_f32_16x16x32_f16(false, af, false, bf,
                                                 (short)0, acc, false, false);
    __syncthreads();
  }
  float* Sb = S + (size_t)b * NN * NN;
#pragma unroll
  for (int r = 0; r < 8; ++r) {
    const int row = n0 + wr * 16 + r + 8 * half;
    const int col = c0 + wc * 16 + idx;
    Sb[(size_t)row * NN + col] = acc[r];
  }
}

// ---------------------------------------------------------------------------
// 4) column softmax over axis n:  S[b,:,c] = softmax(S[b,:,c])
// ---------------------------------------------------------------------------
__global__ void __launch_bounds__(256)
k_softmax(float* __restrict__ S) {
  const int c = blockIdx.x, b = blockIdx.y;
  float* col = S + (size_t)b * NN * NN + c;
  __shared__ float red[256];
  const int tid = threadIdx.x;
  float m = -1e30f;
  for (int n = tid; n < NN; n += 256) m = fmaxf(m, col[(size_t)n * NN]);
  red[tid] = m; __syncthreads();
  for (int s = 128; s > 0; s >>= 1) {
    if (tid < s) red[tid] = fmaxf(red[tid], red[tid + s]);
    __syncthreads();
  }
  m = red[0]; __syncthreads();
  float sum = 0.f;
  for (int n = tid; n < NN; n += 256) {
    const float e = expf(col[(size_t)n * NN] - m);
    col[(size_t)n * NN] = e;
    sum += e;
  }
  red[tid] = sum; __syncthreads();
  for (int s = 128; s > 0; s >>= 1) {
    if (tid < s) red[tid] += red[tid + s];
    __syncthreads();
  }
  const float inv = 1.f / red[0];
  for (int n = tid; n < NN; n += 256) col[(size_t)n * NN] *= inv;
}

// ---------------------------------------------------------------------------
// 5) Z[b,n,c] = sum_m (cheb_k[m,n]*S[b,m,n]) * x[b,m,f,t], c = f*T+t
//    K = 512; cheb∘S fused into A staging; LDS-staged f16 operands.
// ---------------------------------------------------------------------------
__global__ void __launch_bounds__(256)
k_Z(const float* __restrict__ chebk, const float* __restrict__ S,
    const float* __restrict__ x, float* __restrict__ Z) {
  const int c0 = blockIdx.x * 32, n0 = blockIdx.y * 64, b = blockIdx.z;
  const int tid  = threadIdx.x;
  const int wave = tid >> 5, lane = tid & 31;
  const int half = lane >> 4, idx = lane & 15;
  const int wr = wave >> 1, wc = wave & 1;

  __shared__ _Float16 sA[64][32];                // [m][k]
  __shared__ _Float16 sBt[32][32];               // [n][k]

  const float* Sb = S + (size_t)b * NN * NN;
  const float* xb = x + (size_t)b * NN * FIN * TT;
  v8f acc = {};

  for (int k0 = 0; k0 < NN; k0 += 32) {
    // Stage A = cheb_k ∘ S, transposed: source rows are k-major, m contiguous
    for (int it = tid; it < 32 * 64; it += 256) {
      const int kk = it >> 6, m = it & 63;
      const size_t o = (size_t)(k0 + kk) * NN + n0 + m;
      sA[m][kk] = (_Float16)(chebk[o] * Sb[o]);
      if (k0 + 32 < NN) __builtin_prefetch(Sb + o + (size_t)32 * NN, 0, 1);
    }
    // Stage B^T: x rows k-major, c contiguous (c maps linearly to f*T+t)
    for (int it = tid; it < 32 * 32; it += 256) {
      const int k = it >> 5, n = it & 31;
      const float* src = xb + (size_t)(k0 + k) * (FIN * TT) + c0 + n;
      sBt[n][k] = (_Float16)(*src);
      if (k0 + 32 < NN) __builtin_prefetch(src + (size_t)32 * FIN * TT, 0, 1);
    }
    __syncthreads();
    const v16h af = frag_a_lds(&sA[wr * 16 + idx][0], half);
    const v16h bf = *reinterpret_cast<const v16h*>(&sBt[wc * 16 + idx][16 * half]);
    acc = __builtin_amdgcn_wmma_f32_16x16x32_f16(false, af, false, bf,
                                                 (short)0, acc, false, false);
    __syncthreads();
  }
#pragma unroll
  for (int r = 0; r < 8; ++r) {
    const int row = n0 + wr * 16 + r + 8 * half;
    const int col = c0 + wc * 16 + idx;
    Z[((size_t)b * NN + row) * (FIN * TT) + col] = acc[r];
  }
}

// ---------------------------------------------------------------------------
// 6) G[b,n,o,t] += sum_f Z[b,n,f,t] * Theta_k[f,o]   (K = 64, tiny operands)
// ---------------------------------------------------------------------------
__global__ void __launch_bounds__(32)
k_theta(const float* __restrict__ Z, const float* __restrict__ Thk,
        float* __restrict__ G) {
  const int t0 = blockIdx.x * 16, o0 = blockIdx.y * 16;
  const size_t r = blockIdx.z;                     // r = b*N + n
  const float* Zr = Z + r * (FIN * TT);
  v8f acc = {};
  acc = wmma_tile_f16(
      [&](int m, int f) { return Thk[f * OO + (o0 + m)]; },
      [&](int f, int n) {
        const int t = t0 + n;
        return (t < TT) ? Zr[f * TT + t] : 0.f;
      },
      FIN, acc);
  const int lane = threadIdx.x & 31, half = lane >> 4, idx = lane & 15;
#pragma unroll
  for (int rr = 0; rr < 8; ++rr) {
    const int o = o0 + rr + 8 * half, t = t0 + idx;
    if (t < TT) G[(r * OO + o) * TT + t] += acc[rr];
  }
}

// ---------------------------------------------------------------------------
// 7) Fully fused Mamba block: one 128-thread block per row r = b*N+n.
//    relu(G) -> LN -> Win -> depthwise causal conv4 + silu -> Wxp ->
//    softplus(dt@Wdt+bdt) -> 24-step SSM scan (state in registers) ->
//    silu(z) gate -> Wout + residual -> M[b,n,o,t]
// ---------------------------------------------------------------------------
__global__ void __launch_bounds__(128)
k_mamba(const float* __restrict__ G, const float* __restrict__ mg,
        const float* __restrict__ mb, const float* __restrict__ Win,
        const float* __restrict__ conv_w, const float* __restrict__ conv_b,
        const float* __restrict__ Wxp, const float* __restrict__ Wdt,
        const float* __restrict__ bdt, const float* __restrict__ A_log,
        const float* __restrict__ Dp, const float* __restrict__ Wout,
        float* __restrict__ M) {
  const size_t r = blockIdx.x;
  const int tid = threadIdx.x;
  __shared__ float s_xin[TT][OO];
  __shared__ float s_xup[TT][DI];
  __shared__ float s_xu[TT][DI];
  __shared__ float s_z[TT][DI];
  __shared__ float s_xdbl[TT][DTR + 2 * DS];
  __shared__ float s_y[TT][DI];
  __shared__ float s_mu[TT], s_rs[TT];

  const float* Gr = G + r * OO * TT;
  for (int i = tid; i < TT * OO; i += 128) {
    const int t = i / OO, o = i - t * OO;
    const float v = Gr[o * TT + t];
    s_xin[t][o] = v > 0.f ? v : 0.f;
  }
  __syncthreads();
  if (tid < TT) {
    float mu = 0.f;
    for (int o = 0; o < OO; ++o) mu += s_xin[tid][o];
    mu *= (1.f / OO);
    float var = 0.f;
    for (int o = 0; o < OO; ++o) { const float d = s_xin[tid][o] - mu; var += d * d; }
    var *= (1.f / OO);
    s_mu[tid] = mu;
    s_rs[tid] = rsqrtf(var + EPSL);
  }
  __syncthreads();
  for (int i = tid; i < TT * 2 * DI; i += 128) {
    const int t = i / (2 * DI), j = i - t * (2 * DI);
    const float mu = s_mu[t], rs = s_rs[t];
    const float* wr = Win + j * OO;
    float acc = 0.f;
    for (int o = 0; o < OO; ++o) {
      const float h = (s_xin[t][o] - mu) * rs * mg[o] + mb[o];
      acc += h * wr[o];
    }
    if (j < DI) s_xup[t][j] = acc; else s_z[t][j - DI] = acc;
  }
  __syncthreads();
  for (int i = tid; i < TT * DI; i += 128) {
    const int t = i / DI, d = i - t * DI;
    float acc = conv_b[d];
#pragma unroll
    for (int w = 0; w < DCONVW; ++w) {
      const int tt = t - (DCONVW - 1) + w;
      if (tt >= 0) acc += conv_w[d * DCONVW + w] * s_xup[tt][d];
    }
    s_xu[t][d] = acc / (1.f + expf(-acc));
  }
  __syncthreads();
  for (int i = tid; i < TT * (DTR + 2 * DS); i += 128) {
    const int t = i / (DTR + 2 * DS), j = i - t * (DTR + 2 * DS);
    const float* wr = Wxp + j * DI;
    float acc = 0.f;
    for (int d = 0; d < DI; ++d) acc += s_xu[t][d] * wr[d];
    s_xdbl[t][j] = acc;
  }
  __syncthreads();
  {
    const int d = tid;
    float negA[DS];
#pragma unroll
    for (int s = 0; s < DS; ++s) negA[s] = -expf(A_log[d * DS + s]);
    const float w0 = Wdt[d * 4 + 0], w1 = Wdt[d * 4 + 1];
    const float w2 = Wdt[d * 4 + 2], w3 = Wdt[d * 4 + 3];
    const float bd = bdt[d], dpd = Dp[d];
    float h[DS];
#pragma unroll
    for (int s = 0; s < DS; ++s) h[s] = 0.f;
    for (int t = 0; t < TT; ++t) {
      const float dtv = s_xdbl[t][0] * w0 + s_xdbl[t][1] * w1 +
                        s_xdbl[t][2] * w2 + s_xdbl[t][3] * w3 + bd;
      const float delta = (dtv > 20.f) ? dtv : log1pf(expf(dtv));
      const float u = s_xu[t][d];
      const float du = delta * u;
      float y = 0.f;
#pragma unroll
      for (int s = 0; s < DS; ++s) {
        h[s] = expf(delta * negA[s]) * h[s] + du * s_xdbl[t][DTR + s];
        y += h[s] * s_xdbl[t][DTR + DS + s];
      }
      y += u * dpd;
      const float zv = s_z[t][d];
      y *= zv / (1.f + expf(-zv));
      s_y[t][d] = y;
    }
  }
  __syncthreads();
  float* Mr = M + r * OO * TT;
  for (int i = tid; i < TT * OO; i += 128) {
    const int t = i / OO, o = i - t * OO;
    const float* wr = Wout + o * DI;
    float acc = s_xin[t][o];
    for (int d = 0; d < DI; ++d) acc += s_y[t][d] * wr[d];
    Mr[o * TT + t] = acc;
  }
}

// ---------------------------------------------------------------------------
// 8) out = relu(LN_over_O( x@Wres.T + bres + x_mamba )), stored [b,n,o,t]
// ---------------------------------------------------------------------------
__global__ void __launch_bounds__(64)
k_final(const float* __restrict__ x, const float* __restrict__ Wres,
        const float* __restrict__ bres, const float* __restrict__ M,
        const float* __restrict__ ln_g, const float* __restrict__ ln_b,
        float* __restrict__ out) {
  const int t = blockIdx.x, n = blockIdx.y, b = blockIdx.z;
  const int o = threadIdx.x;                // 0..63
  __shared__ float s_v[OO];
  __shared__ float s_stat[2];
  const float* xr = x + ((size_t)b * NN + n) * FIN * TT;
  const float* wr = Wres + o * FIN;
  float acc = bres[o];
  for (int f = 0; f < FIN; ++f) acc += wr[f] * xr[f * TT + t];
  acc += M[(((size_t)b * NN + n) * OO + o) * TT + t];
  s_v[o] = acc;
  __syncthreads();
  if (o == 0) {
    float mu = 0.f;
    for (int i = 0; i < OO; ++i) mu += s_v[i];
    mu *= (1.f / OO);
    float var = 0.f;
    for (int i = 0; i < OO; ++i) { const float d = s_v[i] - mu; var += d * d; }
    var *= (1.f / OO);
    s_stat[0] = mu;
    s_stat[1] = rsqrtf(var + EPSL);
  }
  __syncthreads();
  const float v = (acc - s_stat[0]) * s_stat[1] * ln_g[o] + ln_b[o];
  out[(((size_t)b * NN + n) * OO + o) * TT + t] = v > 0.f ? v : 0.f;
}

// ---------------------------------------------------------------------------
extern "C" void kernel_launch(void* const* d_in, const int* in_sizes, int n_in,
                              void* d_out, int out_size, void* d_ws, size_t ws_size,
                              hipStream_t stream) {
  (void)in_sizes; (void)n_in; (void)out_size; (void)ws_size;
  const float* x      = (const float*)d_in[0];
  const float* cheb   = (const float*)d_in[1];
  const float* W1     = (const float*)d_in[2];
  const float* W2     = (const float*)d_in[3];
  const float* W3     = (const float*)d_in[4];
  const float* bsp    = (const float*)d_in[5];
  const float* Vs     = (const float*)d_in[6];
  const float* Theta  = (const float*)d_in[7];
  const float* Wres   = (const float*)d_in[8];
  const float* bres   = (const float*)d_in[9];
  const float* mg     = (const float*)d_in[10];
  const float* mb     = (const float*)d_in[11];
  const float* Win    = (const float*)d_in[12];
  const float* conv_w = (const float*)d_in[13];
  const float* conv_b = (const float*)d_in[14];
  const float* Wxp    = (const float*)d_in[15];
  const float* Wdt    = (const float*)d_in[16];
  const float* bdt    = (const float*)d_in[17];
  const float* A_log  = (const float*)d_in[18];
  const float* Dp     = (const float*)d_in[19];
  const float* Wout   = (const float*)d_in[20];
  const float* ln_g   = (const float*)d_in[21];
  const float* ln_b   = (const float*)d_in[22];
  float* out = (float*)d_out;

  // Workspace layout (floats); Z buffer is reused for the mamba output.
  float* ws = (float*)d_ws;
  size_t off = 0;
  float* sig = ws + off; off += (size_t)BB * NN * NN;
  float* Sb  = ws + off; off += (size_t)BB * NN * NN;
  float* Zb  = ws + off; off += (size_t)BB * NN * FIN * TT;
  float* Gb  = ws + off; off += (size_t)BB * NN * OO * TT;
  float* lhs = ws + off; off += (size_t)BB * NN * TT;
  float* rhs = ws + off; off += (size_t)BB * NN * TT;
  float* Mb  = Zb;  // safe: GCN stage fully done before k_mamba writes here

  k_lhs_rhs<<<dim3(NN, BB), 64, 0, stream>>>(x, W1, W2, W3, lhs, rhs);
  k_prod_sig<<<dim3(NN / 16, NN / 16, BB), 32, 0, stream>>>(lhs, rhs, bsp, sig);
  k_Sgemm<<<dim3(NN / 32, NN / 64, BB), 256, 0, stream>>>(Vs, sig, Sb);
  k_softmax<<<dim3(NN, BB), 256, 0, stream>>>(Sb);
  hipMemsetAsync(Gb, 0, (size_t)BB * NN * OO * TT * sizeof(float), stream);
  for (int k = 0; k < KCH; ++k) {
    k_Z<<<dim3(FIN * TT / 32, NN / 64, BB), 256, 0, stream>>>(
        cheb + (size_t)k * NN * NN, Sb, x, Zb);
    k_theta<<<dim3(2, OO / 16, BB * NN), 32, 0, stream>>>(
        Zb, Theta + (size_t)k * FIN * OO, Gb);
  }
  k_mamba<<<BB * NN, 128, 0, stream>>>(Gb, mg, mb, Win, conv_w, conv_b,
                                       Wxp, Wdt, bdt, A_log, Dp, Wout, Mb);
  k_final<<<dim3(TT, NN, BB), 64, 0, stream>>>(x, Wres, bres, Mb, ln_g, ln_b, out);
}
